// LlamaQJLAttention_46119358825138
// MI455X (gfx1250) — matrix operand order repaired and checked by reading, not compiled, not run on hardware
//
#include <hip/hip_runtime.h>
#include <hip/hip_bf16.h>

typedef __attribute__((ext_vector_type(16))) __bf16 v16bf;
typedef __attribute__((ext_vector_type(8)))  float  v8f;

#define B_   4
#define TP_  8192
#define T_   (TP_ + 1)
#define E_   4096
#define H_   32
#define HKV_ 8
#define G_   4
#define D_   128
#define M_   256
#define NEG_BIG (-3.4028234663852886e38f)

// ---------------- workspace layout (float-unit offsets) ----------------
#define OFF_Q     0            // B*H*D      = 16384
#define OFF_KN    16384        // B*HKV*D    = 4096
#define OFF_VN    20480        // 4096
#define OFF_CHS   24576        // 4096
#define OFF_OM    28672        // 4096
#define OFF_QS    32768        // B*HKV*G*M  = 32768
#define OFF_QO    65536        // B*HKV*G*D  = 16384
#define OFF_ATTN  81920        // B*HKV*G*T  = 1048704
#define OFF_ATTNO 1130624      // 16384
#define OFF_SSWZ  1147008      // 32768 bf16 = 16384 float slots
#define OFF_BFRG  1163392      // 32*12*32*16 bf16 = 98304 float slots
// total ~1261696 floats (~4.9 MB)

// ---------------- K1: fused QKV projection (GEMV, wave per row) ----------------
__global__ __launch_bounds__(256) void qkv_kernel(
    const float* __restrict__ hidden, const float* __restrict__ Wq,
    const float* __restrict__ Wk, const float* __restrict__ Wv,
    float* __restrict__ q, float* __restrict__ kn, float* __restrict__ vn)
{
    const int wave = (blockIdx.x * blockDim.x + threadIdx.x) >> 5;
    const int lane = threadIdx.x & 31;
    if (wave >= H_ * D_ + 2 * HKV_ * D_) return;
    const float* wrow;
    if (wave < H_ * D_)                 wrow = Wq + (size_t)wave * E_;
    else if (wave < H_*D_ + HKV_*D_)    wrow = Wk + (size_t)(wave - H_*D_) * E_;
    else                                wrow = Wv + (size_t)(wave - H_*D_ - HKV_*D_) * E_;

    float acc[4] = {0.f, 0.f, 0.f, 0.f};
    for (int j = 0; j < E_; j += 128) {
        const float4 w4 = *(const float4*)(wrow + j + lane * 4);
#pragma unroll
        for (int b = 0; b < 4; ++b) {
            const float4 h4 = *(const float4*)(hidden + b * E_ + j + lane * 4);
            acc[b] += w4.x * h4.x + w4.y * h4.y + w4.z * h4.z + w4.w * h4.w;
        }
    }
#pragma unroll
    for (int b = 0; b < 4; ++b)
        for (int off = 16; off > 0; off >>= 1) acc[b] += __shfl_xor(acc[b], off);
    if (lane == 0) {
        if (wave < H_ * D_) {
            int h = wave >> 7, d = wave & 127;
            for (int b = 0; b < 4; ++b) q[(b * H_ + h) * D_ + d] = acc[b];
        } else if (wave < H_*D_ + HKV_*D_) {
            int r = wave - H_*D_; int h = r >> 7, d = r & 127;
            for (int b = 0; b < 4; ++b) kn[(b * HKV_ + h) * D_ + d] = acc[b];
        } else {
            int r = wave - H_*D_ - HKV_*D_; int h = r >> 7, d = r & 127;
            for (int b = 0; b < 4; ++b) vn[(b * HKV_ + h) * D_ + d] = acc[b];
        }
    }
}

// ---------------- K2: RoPE on q and k_new (thread per rotation pair) ----------------
__global__ void rope_kernel(const float* __restrict__ cosb, const float* __restrict__ sinb,
                            float* __restrict__ q, float* __restrict__ kn)
{
    int tid = blockIdx.x * blockDim.x + threadIdx.x;
    if (tid >= B_ * (H_ + HKV_) * 64) return;
    int i  = tid & 63;
    int hh = (tid >> 6) % (H_ + HKV_);
    int b  = tid / (64 * (H_ + HKV_));
    float* p = (hh < H_) ? (q + ((size_t)(b * H_ + hh)) * D_)
                         : (kn + ((size_t)(b * HKV_ + (hh - H_))) * D_);
    const float* c = cosb + b * D_;
    const float* s = sinb + b * D_;
    float x0 = p[i], x1 = p[i + 64];
    p[i]      = x0 * c[i]      - x1 * s[i];
    p[i + 64] = x1 * c[i + 64] + x0 * s[i + 64];
}

// ---------------- K3: pre-swizzle S_proj into WMMA B-fragment order (bf16) ---------
// layout: [c(4)][ntile(16)][lane(32)][h(16)]; B[k][n]: k = 32c + h + 16*(lane>=16),
// n-col = lane&15, global m = ntile*16 + (lane&15)
__global__ void sswz_kernel(const float* __restrict__ S, __bf16* __restrict__ sswz)
{
    int e = blockIdx.x * blockDim.x + threadIdx.x;
    if (e >= 4 * 16 * 32 * 16) return;
    int hh = e & 15, lane = (e >> 4) & 31, nt = (e >> 9) & 15, c = e >> 13;
    int k = 32 * c + hh + ((lane >= 16) ? 16 : 0);
    int m = nt * 16 + (lane & 15);
    sswz[e] = (__bf16)S[k * M_ + m];
}

// ---------------- K4: channel score = mean |keys| over T ----------------
__global__ __launch_bounds__(256) void chscore_kernel(
    const float* __restrict__ pk, const float* __restrict__ kn, float* __restrict__ chs)
{
    const int bk = blockIdx.x, tid = threadIdx.x;
    const int d = tid & 127, ph = tid >> 7;
    const float* base = pk + (size_t)bk * TP_ * D_;
    float acc = 0.f;
    for (int t = ph; t < TP_; t += 2) acc += fabsf(base[(size_t)t * D_ + d]);
    __shared__ float red[256];
    red[tid] = acc;
    __syncthreads();
    if (ph == 0)
        chs[bk * D_ + d] = (red[d] + red[d + 128] + fabsf(kn[bk * D_ + d])) / (float)T_;
}

// ---------------- K5: top-8 channels -> outlier mask ----------------
__global__ __launch_bounds__(128) void topk_kernel(const float* __restrict__ chs,
                                                   float* __restrict__ om)
{
    const int bk = blockIdx.x, d = threadIdx.x;
    __shared__ float vals[128];
    __shared__ float v[128];
    __shared__ int   ix[128];
    vals[d] = chs[bk * D_ + d];
    float o = 0.f;
    __syncthreads();
    for (int it = 0; it < 8; ++it) {
        v[d] = vals[d]; ix[d] = d;
        __syncthreads();
        for (int s = 64; s > 0; s >>= 1) {
            if (d < s) {
                if (v[d + s] > v[d] || (v[d + s] == v[d] && ix[d + s] < ix[d])) {
                    v[d] = v[d + s]; ix[d] = ix[d + s];
                }
            }
            __syncthreads();
        }
        if (d == ix[0]) { o = 1.f; vals[d] = NEG_BIG; }
        __syncthreads();
    }
    om[bk * D_ + d] = o;
}

// ---------------- K6: qs = (q_in @ S)*cEst ; q_out = q*omask*cExact ----------------
__global__ __launch_bounds__(256) void qprep_kernel(
    const float* __restrict__ q, const float* __restrict__ om,
    const float* __restrict__ S, float* __restrict__ qs_s, float* __restrict__ qo_s)
{
    const int bkg = blockIdx.x;            // (b,kv,g)
    const int g = bkg & 3, bk = bkg >> 2;
    const int kv = bk & 7, b = bk >> 3;
    const float cExact = 1.0f / sqrtf((float)D_);
    const float cEst = sqrtf(3.14159265358979323846f * 0.5f) / (float)M_ * cExact;
    const float* qrow = q + ((size_t)(b * H_ + kv * G_ + g)) * D_;
    const float* omr  = om + bk * D_;
    __shared__ float qin[128];
    const int tid = threadIdx.x;
    if (tid < 128) {
        float qq = qrow[tid], o = omr[tid];
        qin[tid] = qq * (1.f - o);
        qo_s[bkg * D_ + tid] = qq * o * cExact;
    }
    __syncthreads();
    float acc = 0.f;
    for (int d = 0; d < 128; ++d) acc += qin[d] * S[d * M_ + tid];
    qs_s[bkg * M_ + tid] = acc * cEst;
}

// ---------------- K7: pack [qs ; q_out] B-fragments (K=384 x N=16) bf16 ----------------
__global__ __launch_bounds__(256) void bfrag_kernel(
    const float* __restrict__ qs_s, const float* __restrict__ qo_s, __bf16* __restrict__ bf)
{
    const int bk = blockIdx.x;
    for (int e = threadIdx.x; e < 12 * 32 * 16; e += 256) {
        int hh = e & 15, lane = (e >> 4) & 31, c2 = e >> 9;
        int n = lane & 15;
        int k = 32 * c2 + hh + ((lane >= 16) ? 16 : 0);
        float vv = 0.f;
        if (n < 4) {
            if (k < 256) vv = qs_s[(bk * 4 + n) * M_ + k];
            else         vv = qo_s[(bk * 4 + n) * D_ + (k - 256)];
        }
        bf[(size_t)bk * (12 * 32 * 16) + e] = (__bf16)vv;
    }
}

// ---------------- K8: WMMA score kernel ----------------
// One wave per 16-token tile: P = k_in(16x128) @ S(128x256) via 64 bf16 WMMAs,
// then attn(16x16, g<4 valid) = [sign(P)*norm | keys_bf16](16x384) @ [qs;q_out](384x16)
// via 12 bf16 WMMAs. est/exact scaling pre-folded into qs/q_out.
__global__ __launch_bounds__(64) void score_kernel(
    const float* __restrict__ pk, const float* __restrict__ kn,
    const float* __restrict__ om, const __bf16* __restrict__ sswz,
    const __bf16* __restrict__ bfr, const float* __restrict__ mask,
    float* __restrict__ attn)
{
    const int bk = blockIdx.y;
    const int w = threadIdx.x >> 5;
    const int lane = threadIdx.x & 31;
    const int t0 = (blockIdx.x * 2 + w) * 16;

    __shared__ __align__(16) float  keysF[2][16 * 128];   // 16 KB
    __shared__ __bf16               signsS[2][16 * 256];  // 16 KB
    __shared__ float                omS[128];
    __shared__ float                normS[2][16];

    for (int i = threadIdx.x; i < 128; i += 64) omS[i] = om[bk * D_ + i];

    const float* kbase = pk + (size_t)bk * TP_ * D_;
#pragma unroll 4
    for (int tok = 0; tok < 16; ++tok) {
        int t = t0 + tok;
        float4 val;
        if (t < TP_)       val = *(const float4*)(kbase + (size_t)t * D_ + lane * 4);
        else if (t == TP_) val = *(const float4*)(kn + bk * D_ + lane * 4);
        else               val = make_float4(0.f, 0.f, 0.f, 0.f);
        *(float4*)(&keysF[w][tok * 128 + lane * 4]) = val;
        if (t + 16 < TP_) // stream-prefetch next tile's rows (global_prefetch_b8)
            __builtin_prefetch(kbase + (size_t)(t + 16) * D_ + lane * 4, 0, 0);
    }
    __syncthreads();

    // per-token norms of masked keys (lanes 0..15 of each wave)
    if (lane < 16) {
        float s = 0.f;
        for (int d = 0; d < 128; ++d) {
            float x = keysF[w][lane * 128 + d] * (1.f - omS[d]);
            s += x * x;
        }
        normS[w][lane] = sqrtf(s);
    }
    __syncthreads();

    // A fragments of masked keys for the sketch GEMM (4 K-chunks of 32)
    const int row  = lane & 15;
    const int kofA = (lane >= 16) ? 16 : 0;
    v16bf afr[4];
#pragma unroll
    for (int c = 0; c < 4; ++c)
#pragma unroll
        for (int hh = 0; hh < 16; ++hh) {
            int kd = 32 * c + hh + kofA;
            afr[c][hh] = (__bf16)(keysF[w][row * 128 + kd] * (1.f - omS[kd]));
        }

    // P = k_in @ S ; write sign(P)*norm to LDS in (token, m) order
    const int tokb = (lane >= 16) ? 8 : 0;
    for (int nt = 0; nt < 16; ++nt) {
        v8f acc = {};
#pragma unroll
        for (int c = 0; c < 4; ++c) {
            v16bf bfrg = *(const v16bf*)(sswz + ((size_t)(c * 16 + nt) * 32 + lane) * 16);
            acc = __builtin_amdgcn_wmma_f32_16x16x32_bf16(false, afr[c], false, bfrg,
                                                          (short)0, acc, false, false);
        }
        int m = nt * 16 + (lane & 15);
#pragma unroll
        for (int r = 0; r < 8; ++r) {
            int tok = r + tokb;
            float p = acc[r];
            float nv = normS[w][tok];
            float sv = (p > 0.f) ? nv : ((p < 0.f) ? -nv : 0.f);
            signsS[w][tok * 256 + m] = (__bf16)sv;
        }
    }
    __syncthreads();

    // fused est+exact GEMM: K = 256 (signs*norm vs qs) + 128 (keys vs q_out)
    v8f acc2 = {};
#pragma unroll
    for (int c2 = 0; c2 < 12; ++c2) {
        v16bf a2;
#pragma unroll
        for (int hh = 0; hh < 16; ++hh) {
            int kk = 32 * c2 + hh + kofA;
            a2[hh] = (kk < 256) ? signsS[w][row * 256 + kk]
                                : (__bf16)keysF[w][row * 128 + (kk - 256)];
        }
        v16bf b2 = *(const v16bf*)(bfr + ((size_t)bk * 12 + c2) * 32 * 16 + lane * 16);
        acc2 = __builtin_amdgcn_wmma_f32_16x16x32_bf16(false, a2, false, b2,
                                                       (short)0, acc2, false, false);
    }

    const int g = lane & 15;
    if (g < 4) {
        const int b = bk >> 3;
#pragma unroll
        for (int r = 0; r < 8; ++r) {
            int t = t0 + r + tokb;
            if (t < T_) {
                float mv = mask[b * T_ + t];
                attn[((size_t)(bk * 4 + g)) * T_ + t] = acc2[r] + (1.f - mv) * NEG_BIG;
            }
        }
    }
}

// ---------------- K9: softmax over T + weighted values (values read once per (b,kv)) ----
__global__ __launch_bounds__(256) void softmax_av_kernel(
    const float* __restrict__ attn, const float* __restrict__ pv,
    const float* __restrict__ vn, float* __restrict__ attno)
{
    const int bk = blockIdx.x, tid = threadIdx.x;
    __shared__ float red[256];
    __shared__ float gmax[4], gsum[4];
    __shared__ float wbuf[64][4];
    __shared__ float accS[2][4][128];

    for (int g = 0; g < 4; ++g) {
        const float* arow = attn + (size_t)(bk * 4 + g) * T_;
        float m = NEG_BIG;
        for (int t = tid; t < T_; t += 256) m = fmaxf(m, arow[t]);
        red[tid] = m; __syncthreads();
        for (int s = 128; s > 0; s >>= 1) { if (tid < s) red[tid] = fmaxf(red[tid], red[tid + s]); __syncthreads(); }
        if (tid == 0) gmax[g] = red[0];
        __syncthreads();
        float sm = 0.f;
        for (int t = tid; t < T_; t += 256) sm += __expf(arow[t] - gmax[g]);
        red[tid] = sm; __syncthreads();
        for (int s = 128; s > 0; s >>= 1) { if (tid < s) red[tid] += red[tid + s]; __syncthreads(); }
        if (tid == 0) gsum[g] = red[0];
        __syncthreads();
    }

    const int d = tid & 127, ph = tid >> 7;
    const float* vbase = pv + (size_t)bk * TP_ * D_;
    float acc[4] = {0.f, 0.f, 0.f, 0.f};
    for (int tb = 0; tb < T_; tb += 64) {
        int tt = tb + (tid >> 2), g2 = tid & 3;
        float wv = 0.f;
        if (tt < T_) wv = __expf(attn[(size_t)(bk * 4 + g2) * T_ + tt] - gmax[g2]);
        wbuf[tid >> 2][g2] = wv;
        __syncthreads();
        int pft = (tb + 64 < TP_) ? (tb + 64) : (TP_ - 1);
        __builtin_prefetch(vbase + (size_t)pft * D_ + d, 0, 0);
        for (int i = ph; i < 64; i += 2) {
            int t = tb + i;
            if (t >= T_) break;
            float v = (t < TP_) ? vbase[(size_t)t * D_ + d] : vn[bk * D_ + d];
            acc[0] += wbuf[i][0] * v; acc[1] += wbuf[i][1] * v;
            acc[2] += wbuf[i][2] * v; acc[3] += wbuf[i][3] * v;
        }
        __syncthreads();
    }
#pragma unroll
    for (int g = 0; g < 4; ++g) accS[ph][g][d] = acc[g];
    __syncthreads();
    if (ph == 0) {
        const int b = bk >> 3, kv = bk & 7;
        for (int g = 0; g < 4; ++g)
            attno[((size_t)(b * H_ + kv * G_ + g)) * D_ + d] =
                (accS[0][g][d] + accS[1][g][d]) / gsum[g];
    }
}

// ---------------- K10: output projection (wave per output column, 4 batches shared) ----
__global__ __launch_bounds__(256) void outproj_kernel(
    const float* __restrict__ attno, const float* __restrict__ Wo, float* __restrict__ out)
{
    const int wave = (blockIdx.x * blockDim.x + threadIdx.x) >> 5;
    const int lane = threadIdx.x & 31;
    if (wave >= E_) return;
    const float* wrow = Wo + (size_t)wave * (H_ * D_);
    float acc[4] = {0.f, 0.f, 0.f, 0.f};
    for (int j = 0; j < H_ * D_; j += 128) {
        const float4 w4 = *(const float4*)(wrow + j + lane * 4);
#pragma unroll
        for (int b = 0; b < 4; ++b) {
            const float4 h4 = *(const float4*)(attno + b * (H_ * D_) + j + lane * 4);
            acc[b] += w4.x * h4.x + w4.y * h4.y + w4.z * h4.z + w4.w * h4.w;
        }
    }
#pragma unroll
    for (int b = 0; b < 4; ++b)
        for (int off = 16; off > 0; off >>= 1) acc[b] += __shfl_xor(acc[b], off);
    if (lane == 0)
        for (int b = 0; b < 4; ++b) out[b * E_ + wave] = acc[b];
}

// ---------------- launch ----------------
extern "C" void kernel_launch(void* const* d_in, const int* in_sizes, int n_in,
                              void* d_out, int out_size, void* d_ws, size_t ws_size,
                              hipStream_t stream)
{
    const float* hidden = (const float*)d_in[0];
    const float* cosb   = (const float*)d_in[1];
    const float* sinb   = (const float*)d_in[2];
    const float* pk     = (const float*)d_in[3];
    const float* pv     = (const float*)d_in[4];
    const float* mask   = (const float*)d_in[5];
    const float* Wq     = (const float*)d_in[6];
    const float* Wk     = (const float*)d_in[7];
    const float* Wv     = (const float*)d_in[8];
    const float* Wo     = (const float*)d_in[9];
    const float* S      = (const float*)d_in[10];

    float* ws = (float*)d_ws;
    float* q     = ws + OFF_Q;
    float* kn    = ws + OFF_KN;
    float* vn    = ws + OFF_VN;
    float* chs   = ws + OFF_CHS;
    float* om    = ws + OFF_OM;
    float* qs_s  = ws + OFF_QS;
    float* qo_s  = ws + OFF_QO;
    float* attn  = ws + OFF_ATTN;
    float* attno = ws + OFF_ATTNO;
    __bf16* sswz = (__bf16*)(ws + OFF_SSWZ);
    __bf16* bfr  = (__bf16*)(ws + OFF_BFRG);
    float* out = (float*)d_out;

    qkv_kernel<<<768, 256, 0, stream>>>(hidden, Wq, Wk, Wv, q, kn, vn);
    rope_kernel<<<40, 256, 0, stream>>>(cosb, sinb, q, kn);
    sswz_kernel<<<128, 256, 0, stream>>>(S, sswz);
    chscore_kernel<<<B_ * HKV_, 256, 0, stream>>>(pk, kn, chs);
    topk_kernel<<<B_ * HKV_, 128, 0, stream>>>(chs, om);
    qprep_kernel<<<B_ * HKV_ * G_, 256, 0, stream>>>(q, om, S, qs_s, qo_s);
    bfrag_kernel<<<B_ * HKV_, 256, 0, stream>>>(qs_s, qo_s, bfr);
    score_kernel<<<dim3(257, B_ * HKV_), 64, 0, stream>>>(pk, kn, om, sswz, bfr, mask, attn);
    softmax_av_kernel<<<B_ * HKV_, 256, 0, stream>>>(attn, pv, vn, attno);
    outproj_kernel<<<512, 256, 0, stream>>>(attno, Wo, out);
}
